// Top2Gate_24653112279121
// MI455X (gfx1250) — matrix operand order repaired and checked by reading, not compiled, not run on hardware
//
#include <hip/hip_runtime.h>
#include <hip/hip_bf16.h>

#define S_TOKENS 65536
#define D_MODEL  1024
#define N_EXP    64
#define TPB_TOK  128
#define NBLK     (S_TOKENS / TPB_TOK)   // 512
#define THREADS  256
#define KSTEP    32
#define LSTRIDE  (N_EXP + 1)            // logits LDS row stride (conflict-free)

typedef __attribute__((ext_vector_type(16))) __bf16       v16bf;
typedef __attribute__((ext_vector_type(8)))  float        v8f;
typedef __attribute__((ext_vector_type(4)))  float        f32x4;
typedef __attribute__((ext_vector_type(4)))  unsigned int u32x4;

union Frag { v16bf v; u32x4 q[2]; };
union Pk8  { __bf16 h[8]; u32x4 q; };

// Hardware RNE f32->bf16 (v_cvt_*bf16_f32), packed b128 store to LDS.
__device__ __forceinline__ void cvt_store8(__bf16* dst, const float* src) {
  f32x4 a = *(const f32x4*)src;
  f32x4 b = *(const f32x4*)(src + 4);
  Pk8 p;
  p.h[0] = (__bf16)a.x; p.h[1] = (__bf16)a.y; p.h[2] = (__bf16)a.z; p.h[3] = (__bf16)a.w;
  p.h[4] = (__bf16)b.x; p.h[5] = (__bf16)b.y; p.h[6] = (__bf16)b.z; p.h[7] = (__bf16)b.w;
  *(u32x4*)dst = p.q;
}

// ---------------- Phase 1: logits GEMM (WMMA bf16) + softmax/top2 + partials ---
__global__ __launch_bounds__(THREADS)
void moe_gate_phase1(const float* __restrict__ x, const float* __restrict__ wg,
                     float* __restrict__ out,
                     int* __restrict__ idx1g, int* __restrict__ idx2g,
                     int* __restrict__ rank1g, int* __restrict__ rank2g,
                     int* __restrict__ cnt1g, int* __restrict__ cnt2g,
                     float* __restrict__ mepg) {
  __shared__ __bf16 lds_x[TPB_TOK * KSTEP];   // 8 KB
  __shared__ __bf16 lds_w[N_EXP  * KSTEP];    // 4 KB
  __shared__ float lds_logits[TPB_TOK * LSTRIDE];     // 33.3 KB (logits, then exp(v-max))
  __shared__ float lds_b1[TPB_TOK];
  __shared__ float lds_inv[TPB_TOK];
  __shared__ int   lds_i1[TPB_TOK], lds_i2[TPB_TOK];
  __shared__ int   lds_c1[N_EXP],  lds_c2[N_EXP];
  __shared__ float lds_me4[4 * N_EXP];

  const int tid  = threadIdx.x;
  const int blk  = blockIdx.x;
  const int tok0 = blk * TPB_TOK;
  const int lane = tid & 31;
  const int wave = tid >> 5;
  const int lh   = lane >> 4;      // lane half: 0 or 1
  const int lm   = lane & 15;

  if (tid < N_EXP) { lds_c1[tid] = 0; lds_c2[tid] = 0; }

  v8f acc[4];
  #pragma unroll
  for (int e = 0; e < 4; ++e)
    #pragma unroll
    for (int r = 0; r < 8; ++r) acc[e][r] = 0.0f;

  // ---- K loop: stage bf16 tiles to LDS, 8 waves x (16 tok x 64 exp) WMMA ----
  for (int k0 = 0; k0 < D_MODEL; k0 += KSTEP) {
    {   // x tile: 128 rows x 32 cols, 16 f32 per thread
      int row = tid >> 1;
      int c0  = (tid & 1) * 16;
      const float* src = x + (size_t)(tok0 + row) * D_MODEL + k0 + c0;
      __bf16* dst = &lds_x[row * KSTEP + c0];
      cvt_store8(dst, src);
      cvt_store8(dst + 8, src + 8);
      if (k0 + KSTEP < D_MODEL)
        __builtin_prefetch(src + KSTEP, 0, 3);   // global_prefetch_b8
    }
    {   // wg tile: 64 rows x 32 cols, 8 f32 per thread
      int row = tid >> 2;
      int c0  = (tid & 3) * 8;
      cvt_store8(&lds_w[row * KSTEP + c0], wg + (size_t)row * D_MODEL + k0 + c0);
    }
    __syncthreads();

    // A fragment: row M = wave*16 + lm ; lane<16 holds K{0..7,16..23}, lane>=16 K{8..15,24..31}
    Frag a;
    const __bf16* ar = &lds_x[(wave * 16 + lm) * KSTEP];
    a.q[0] = *(const u32x4*)&ar[lh * 8];
    a.q[1] = *(const u32x4*)&ar[16 + lh * 8];

    #pragma unroll
    for (int e4 = 0; e4 < 4; ++e4) {
      // B fragment: col N = e4*16 + lm ; lane<16 holds K{0..7,16..23}, lane>=16 K{8..15,24..31}
      Frag b;
      const __bf16* br = &lds_w[(e4 * 16 + lm) * KSTEP];
      b.q[0] = *(const u32x4*)&br[lh * 8];
      b.q[1] = *(const u32x4*)&br[16 + lh * 8];
      acc[e4] = __builtin_amdgcn_wmma_f32_16x16x32_bf16(
          false, a.v, false, b.v, (short)0, acc[e4], false, false);
    }
    __syncthreads();
  }

  // ---- spill logits to LDS (C/D layout: M = r + 8*lh, N = lm) ----
  #pragma unroll
  for (int e4 = 0; e4 < 4; ++e4)
    #pragma unroll
    for (int r = 0; r < 8; ++r)
      lds_logits[(wave * 16 + r + 8 * lh) * LSTRIDE + (e4 * 16 + lm)] = acc[e4][r];
  __syncthreads();

  // ---- per-token top2 + softmax gates (threads 0..127, one token each) ----
  if (tid < TPB_TOK) {
    float* lrow = &lds_logits[tid * LSTRIDE];
    float b1 = -3.402823466e38f; int i1 = 0;
    #pragma unroll 8
    for (int e = 0; e < N_EXP; ++e) { float v = lrow[e]; if (v > b1) { b1 = v; i1 = e; } }
    float b2 = -3.402823466e38f; int i2 = 0;
    #pragma unroll 8
    for (int e = 0; e < N_EXP; ++e) {
      if (e == i1) continue;
      float v = lrow[e]; if (v > b2) { b2 = v; i2 = e; }
    }
    // exp pass: accumulate softmax denominator AND cache exp(v-max) in place
    float sum = 0.0f;
    #pragma unroll 8
    for (int e = 0; e < N_EXP; ++e) {
      float p = __expf(lrow[e] - b1);
      lrow[e] = p;                        // reuse for me partials (no 2nd expf pass)
      sum += p;
    }
    float inv = 1.0f / sum;
    float g1 = inv;                       // exp(b1-b1)*inv
    float g2 = lrow[i2] * inv;            // exp(b2-b1)*inv
    float denom = fmaxf(g1 + g2, 1.1920929e-7f);
    int t = tok0 + tid;
    out[1 + t]                 = g1 / denom;
    out[1 + S_TOKENS + t]      = g2 / denom;
    out[1 + 2 * S_TOKENS + t]  = (float)i1;
    out[1 + 3 * S_TOKENS + t]  = (float)i2;
    idx1g[t] = i1;  idx2g[t] = i2;
    lds_i1[tid] = i1;  lds_i2[tid] = i2;
    lds_b1[tid] = b1;  lds_inv[tid] = inv;
    atomicAdd(&lds_c1[i1], 1);            // integer: order-independent
    atomicAdd(&lds_c2[i2], 1);
  }
  __syncthreads();

  // ---- deterministic per-expert me partials: thread = expert + 64*part ----
  {
    int e    = tid & (N_EXP - 1);
    int part = tid >> 6;                  // 0..3, 32 tokens each
    float s = 0.0f;
    int t0 = part * 32;
    for (int t = t0; t < t0 + 32; ++t)
      s += lds_logits[t * LSTRIDE + e] * lds_inv[t];   // rows hold exp(v-max)
    lds_me4[part * N_EXP + e] = s;
  }
  // ---- within-block ranks (cumsum of one-hot inside block) ----
  __syncthreads();
  if (tid < TPB_TOK) {
    int i1 = lds_i1[tid], i2 = lds_i2[tid];
    int r1 = 0, r2 = 0;
    for (int u = 0; u < tid; ++u) {
      r1 += (lds_i1[u] == i1);
      r2 += (lds_i2[u] == i2);
    }
    rank1g[tok0 + tid] = r1;
    rank2g[tok0 + tid] = r2;
  }
  if (tid < N_EXP) {
    cnt1g[blk * N_EXP + tid] = lds_c1[tid];
    cnt2g[blk * N_EXP + tid] = lds_c2[tid];
    mepg[blk * N_EXP + tid]  = lds_me4[tid] + lds_me4[N_EXP + tid]
                             + lds_me4[2 * N_EXP + tid] + lds_me4[3 * N_EXP + tid];
  }
}

// ---------------- Phase 2: per-expert scan across blocks + loss -----------------
__global__ __launch_bounds__(64)
void moe_gate_phase2(float* __restrict__ out,
                     int* __restrict__ cnt1g, int* __restrict__ cnt2g,
                     const float* __restrict__ mepg) {
  __shared__ float red[N_EXP];
  int e = threadIdx.x;
  int run1 = 0, run2 = 0;
  float me = 0.0f;
  for (int b = 0; b < NBLK; ++b) {
    int c1 = cnt1g[b * N_EXP + e];
    int c2 = cnt2g[b * N_EXP + e];
    cnt1g[b * N_EXP + e] = run1;          // exclusive prefix -> offsets1
    cnt2g[b * N_EXP + e] = run2;          // exclusive prefix -> offsets2 (pre ce)
    run1 += c1;  run2 += c2;
    me += mepg[b * N_EXP + e];
  }
  // locations2 gets + sum(mask1) = ce[e]
  for (int b = 0; b < NBLK; ++b) cnt2g[b * N_EXP + e] += run1;

  red[e] = me * (float)run1;              // me[e] * ce[e]
  __syncthreads();
  for (int s = 32; s > 0; s >>= 1) {
    if (e < s) red[e] += red[e + s];
    __syncthreads();
  }
  if (e == 0)
    out[0] = red[0] * ((float)N_EXP / ((float)S_TOKENS * (float)S_TOKENS));
}

// ---------------- Phase 3: locations = block offset + within-block rank ---------
__global__ __launch_bounds__(TPB_TOK)
void moe_gate_phase3(float* __restrict__ out,
                     const int* __restrict__ idx1g, const int* __restrict__ idx2g,
                     const int* __restrict__ rank1g, const int* __restrict__ rank2g,
                     const int* __restrict__ offs1g, const int* __restrict__ offs2g) {
  int t   = blockIdx.x * TPB_TOK + threadIdx.x;
  int blk = blockIdx.x;
  int loc1 = offs1g[blk * N_EXP + idx1g[t]] + rank1g[t];
  int loc2 = offs2g[blk * N_EXP + idx2g[t]] + rank2g[t];
  out[1 + 4 * S_TOKENS + t] = (float)loc1;
  out[1 + 5 * S_TOKENS + t] = (float)loc2;
}

extern "C" void kernel_launch(void* const* d_in, const int* in_sizes, int n_in,
                              void* d_out, int out_size, void* d_ws, size_t ws_size,
                              hipStream_t stream) {
  (void)in_sizes; (void)n_in; (void)out_size; (void)ws_size;
  const float* x  = (const float*)d_in[0];   // (65536, 1024) f32
  const float* wg = (const float*)d_in[1];   // (64, 1024) f32
  float* out = (float*)d_out;                // 1 + 6*65536 floats

  int*   idx1  = (int*)d_ws;                 // S
  int*   idx2  = idx1 + S_TOKENS;            // S
  int*   rank1 = idx2 + S_TOKENS;            // S
  int*   rank2 = rank1 + S_TOKENS;           // S
  int*   cnt1  = rank2 + S_TOKENS;           // NBLK*64 (-> offsets1)
  int*   cnt2  = cnt1 + NBLK * N_EXP;        // NBLK*64 (-> offsets2 + ce)
  float* mep   = (float*)(cnt2 + NBLK * N_EXP); // NBLK*64

  moe_gate_phase1<<<dim3(NBLK), dim3(THREADS), 0, stream>>>(
      x, wg, out, idx1, idx2, rank1, rank2, cnt1, cnt2, mep);
  moe_gate_phase2<<<dim3(1), dim3(64), 0, stream>>>(out, cnt1, cnt2, mep);
  moe_gate_phase3<<<dim3(NBLK), dim3(TPB_TOK), 0, stream>>>(
      out, idx1, idx2, rank1, rank2, cnt1, cnt2);
}
